// EncoderBlock_75746043232660
// MI455X (gfx1250) — matrix operand order repaired and checked by reading, not compiled, not run on hardware
//
#include <hip/hip_runtime.h>

// ---------------------------------------------------------------------------
// Shapes (fixed by the reference)
// ---------------------------------------------------------------------------
#define BB   2
#define SS   2048
#define DD   1024
#define HH   16
#define DK   64
#define DFF  4096
#define MTOT (BB * SS)   // 4096 rows

typedef __attribute__((ext_vector_type(16))) __bf16 v16bf;
typedef __attribute__((ext_vector_type(8)))  float  v8f;

union AF {                 // one 16x32 bf16 A/B operand (8 VGPRs)
    v16bf        v;
    unsigned int u[8];
};

#if __has_builtin(__builtin_amdgcn_global_load_async_to_lds_b128) && \
    __has_builtin(__builtin_amdgcn_s_wait_asynccnt)
#define HAVE_ASYNC_LDS 1
typedef __attribute__((ext_vector_type(4))) int i4v;
typedef __attribute__((address_space(1))) i4v* gp_i4;   // global int4*
typedef __attribute__((address_space(3))) i4v* lp_i4;   // LDS int4*
#else
#define HAVE_ASYNC_LDS 0
#endif

static __device__ __forceinline__ unsigned short f2bf(float f) {
    union { float f; unsigned int u; } x; x.f = f;
    unsigned int r = x.u + 0x7FFFu + ((x.u >> 16) & 1u);   // RNE
    return (unsigned short)(r >> 16);
}

static __device__ __forceinline__ v8f wmma_bf16(const AF& a, const AF& b, v8f c) {
    return __builtin_amdgcn_wmma_f32_16x16x32_bf16(
        /*neg_a=*/false, a.v, /*neg_b=*/false, b.v,
        /*c_mod=*/(short)0, c, /*reuse_a=*/false, /*reuse_b=*/false);
}

static __device__ __forceinline__ v8f vzero8() {
    v8f z;
#pragma unroll
    for (int i = 0; i < 8; ++i) z[i] = 0.0f;
    return z;
}

// ---------------------------------------------------------------------------
// 16-lane butterfly reductions. Prefer v_permlane16_b32 (pure VALU) over
// __shfl_xor (which lowers to ds_bpermute and competes for the LDS pipe).
// Selector nibbles implement lane -> lane^m within each 16-lane row.
// ---------------------------------------------------------------------------
#if __has_builtin(__builtin_amdgcn_permlane16)
static __device__ __forceinline__ float plx(float x, unsigned s0, unsigned s1) {
    union { float f; unsigned u; } a, b; a.f = x;
    b.u = __builtin_amdgcn_permlane16(a.u, a.u, s0, s1, false, false);
    return b.f;
}
static __device__ __forceinline__ float rmax16(float x) {
    x = fmaxf(x, plx(x, 0x67452301u, 0xEFCDAB89u));   // xor 1
    x = fmaxf(x, plx(x, 0x54761032u, 0xDCFE98BAu));   // xor 2
    x = fmaxf(x, plx(x, 0x32107654u, 0xBA98FEDCu));   // xor 4
    x = fmaxf(x, plx(x, 0xFEDCBA98u, 0x76543210u));   // xor 8
    return x;
}
static __device__ __forceinline__ float rsum16(float x) {
    x += plx(x, 0x67452301u, 0xEFCDAB89u);
    x += plx(x, 0x54761032u, 0xDCFE98BAu);
    x += plx(x, 0x32107654u, 0xBA98FEDCu);
    x += plx(x, 0xFEDCBA98u, 0x76543210u);
    return x;
}
#else
static __device__ __forceinline__ float rmax16(float x) {
#pragma unroll
    for (int m = 1; m < 16; m <<= 1) x = fmaxf(x, __shfl_xor(x, m, 32));
    return x;
}
static __device__ __forceinline__ float rsum16(float x) {
#pragma unroll
    for (int m = 1; m < 16; m <<= 1) x += __shfl_xor(x, m, 32);
    return x;
}
#endif

// ---------------------------------------------------------------------------
// f32 -> bf16 weight conversion
// ---------------------------------------------------------------------------
__global__ __launch_bounds__(256) void cvt_f32_bf16(const float* __restrict__ in,
                                                    unsigned short* __restrict__ out,
                                                    int n) {
    for (int i = blockIdx.x * 256 + threadIdx.x; i < n; i += gridDim.x * 256)
        out[i] = f2bf(in[i]);
}

// ---------------------------------------------------------------------------
// LayerNorm (ddof=1, (x-mean)/(std+eps), scalar gamma/beta) -> bf16
// ---------------------------------------------------------------------------
__global__ __launch_bounds__(256) void ln_bf16(const float* __restrict__ x,
                                               unsigned short* __restrict__ xn,
                                               const float* __restrict__ gp,
                                               const float* __restrict__ bp) {
    __shared__ float red[256];
    const int row = blockIdx.x;
    const int tid = threadIdx.x;
    const float* xr = x + (size_t)row * DD;

    float v0 = xr[tid], v1 = xr[tid + 256], v2 = xr[tid + 512], v3 = xr[tid + 768];
    red[tid] = v0 + v1 + v2 + v3;
    __syncthreads();
#pragma unroll
    for (int off = 128; off > 0; off >>= 1) {
        if (tid < off) red[tid] += red[tid + off];
        __syncthreads();
    }
    const float mean = red[0] * (1.0f / (float)DD);
    __syncthreads();

    float d0 = v0 - mean, d1 = v1 - mean, d2 = v2 - mean, d3 = v3 - mean;
    red[tid] = d0 * d0 + d1 * d1 + d2 * d2 + d3 * d3;
    __syncthreads();
#pragma unroll
    for (int off = 128; off > 0; off >>= 1) {
        if (tid < off) red[tid] += red[tid + off];
        __syncthreads();
    }
    const float var = red[0] * (1.0f / (float)(DD - 1));   // unbiased (ddof=1)
    const float inv = 1.0f / (sqrtf(var) + 1e-6f);
    const float g = gp[0], be = bp[0];

    unsigned short* o = xn + (size_t)row * DD;
    o[tid]       = f2bf(g * d0 * inv + be);
    o[tid + 256] = f2bf(g * d1 * inv + be);
    o[tid + 512] = f2bf(g * d2 * inv + be);
    o[tid + 768] = f2bf(g * d3 * inv + be);
}

// ---------------------------------------------------------------------------
// Tiled bf16 WMMA GEMM:  C[M,N] = A[M,K] (bf16, row-major) x W[K,N] (bf16, rm)
// Workgroup tile 128x128, 8 waves (2x4), wave tile 64x32, K-step 32.
// A tile staged with gfx1250 async global->LDS copies when available.
// ---------------------------------------------------------------------------
template <int EPI>
__global__ __launch_bounds__(256) void gemm_bf16_wmma(
    const unsigned short* __restrict__ A, const unsigned short* __restrict__ Bw,
    const float* __restrict__ bias, const float* __restrict__ resid,
    float* __restrict__ outF, unsigned short* __restrict__ outB,
    int M, int N, int K) {
    __shared__ alignas(16) unsigned short As[128 * 32];   // [m][k]
    __shared__ alignas(16) unsigned short Bs[128 * 32];   // [n][k]  (transposed)

    const int tid    = threadIdx.x;
    const int lane   = tid & 31;
    const int wave   = tid >> 5;
    const int wm     = wave >> 2;          // 0..1
    const int wn     = wave & 3;           // 0..3
    const int lane16 = lane & 15;
    const int half   = lane >> 4;
    const int mBase  = blockIdx.y * 128;
    const int nBase  = blockIdx.x * 128;

    v8f acc[4][2];
#pragma unroll
    for (int i = 0; i < 4; ++i)
#pragma unroll
        for (int j = 0; j < 2; ++j) acc[i][j] = vzero8();

    const int rA = tid >> 1, hA = tid & 1;          // A staging map
    const int kB = tid >> 3, nB = (tid & 7) * 16;   // B staging map

    for (int kt = 0; kt < K; kt += 32) {
        __syncthreads();
        // ---- stage A tile (128x32): 32B per thread ----
#if HAVE_ASYNC_LDS
        {
            gp_i4 gsrc = (gp_i4)(A + (size_t)(mBase + rA) * K + kt + hA * 16);
            lp_i4 ldst = (lp_i4)(As + rA * 32 + hA * 16);
            __builtin_amdgcn_global_load_async_to_lds_b128(gsrc,     ldst,     0, 0);
            __builtin_amdgcn_global_load_async_to_lds_b128(gsrc + 1, ldst + 1, 0, 0);
        }
#else
        {
            const uint4* src = (const uint4*)(A + (size_t)(mBase + rA) * K + kt + hA * 16);
            uint4* dst = (uint4*)(As + rA * 32 + hA * 16);
            dst[0] = src[0];
            dst[1] = src[1];
        }
#endif
        // ---- stage B tile (32x128) transposed into Bs[n][k] ----
        {
            const unsigned short* src = Bw + (size_t)(kt + kB) * N + nBase + nB;
            unsigned short tmp[16];
            *(uint4*)(tmp)     = ((const uint4*)src)[0];
            *(uint4*)(tmp + 8) = ((const uint4*)src)[1];
#pragma unroll
            for (int j = 0; j < 16; ++j) Bs[(nB + j) * 32 + kB] = tmp[j];
        }
        // ---- prefetch next K-tile while this one is consumed ----
        if (kt + 32 < K) {
            __builtin_prefetch(A + (size_t)(mBase + rA) * K + kt + 32, 0, 1);
            __builtin_prefetch(Bw + (size_t)(kt + 32 + kB) * N + nBase + nB, 0, 1);
        }
#if HAVE_ASYNC_LDS
        __builtin_amdgcn_s_wait_asynccnt(0);
#endif
        __syncthreads();

        // ---- operand fragments per ISA 16-bit layouts ----
        AF a[4], bfr[2];
#pragma unroll
        for (int ms = 0; ms < 4; ++ms) {
            const int row = wm * 64 + ms * 16 + lane16;
#pragma unroll
            for (int v = 0; v < 8; ++v) {
                const int k = ((v & 4) << 2) + half * 8 + (v & 3) * 2;
                a[ms].u[v] = *(const unsigned int*)(As + row * 32 + k);
            }
        }
#pragma unroll
        for (int ns = 0; ns < 2; ++ns) {
            const int col = wn * 32 + ns * 16 + lane16;
#pragma unroll
            for (int v = 0; v < 8; ++v)
                bfr[ns].u[v] = *(const unsigned int*)(Bs + col * 32 + ((v & 4) << 2) + half * 8 + (v & 3) * 2);
        }
#pragma unroll
        for (int ms = 0; ms < 4; ++ms)
#pragma unroll
            for (int ns = 0; ns < 2; ++ns)
                acc[ms][ns] = wmma_bf16(a[ms], bfr[ns], acc[ms][ns]);
    }

    // ---- epilogue ----
#pragma unroll
    for (int ms = 0; ms < 4; ++ms)
#pragma unroll
        for (int ns = 0; ns < 2; ++ns)
#pragma unroll
            for (int v = 0; v < 8; ++v) {
                const int gRow = mBase + wm * 64 + ms * 16 + v + 8 * half;
                const int gCol = nBase + wn * 32 + ns * 16 + lane16;
                float val = acc[ms][ns][v] + bias[gCol];
                if (EPI == 0) {
                    outF[(size_t)gRow * N + gCol] = val + resid[(size_t)gRow * N + gCol];
                } else if (EPI == 1) {
                    val = fmaxf(val, 0.0f);
                    outB[(size_t)gRow * N + gCol] = f2bf(val);
                } else {
                    const int b = gRow >> 11, s = gRow & (SS - 1);   // S = 2048
                    const int h = gCol >> 6,  dk = gCol & (DK - 1);  // DK = 64
                    if (EPI == 2)
                        outB[(((size_t)b * HH + h) * SS + s) * DK + dk] = f2bf(val);
                    else
                        outB[(((size_t)b * HH + h) * DK + dk) * SS + s] = f2bf(val);
                }
            }
}

// ---------------------------------------------------------------------------
// Flash attention: one wave handles 16 query rows of one (b,h).
// Q,K in [B,H,S,DK] bf16; V in [B,H,DK,S] bf16. kv processed 64 at a time
// (16 WMMAs per online-softmax update). Row stats at lane index v
// (row = v + 8*half), replicated across the 16 lanes of each half.
// ---------------------------------------------------------------------------
__global__ __launch_bounds__(128) void flash_attn(
    const unsigned short* __restrict__ q, const unsigned short* __restrict__ k,
    const unsigned short* __restrict__ vT, const int* __restrict__ mask,
    unsigned short* __restrict__ ctx) {
    __shared__ alignas(16) unsigned short Pls[4 * 16 * 64];   // per-wave P tile

    const int tid    = threadIdx.x;
    const int lane   = tid & 31;
    const int w      = tid >> 5;
    const int lane16 = lane & 15;
    const int half   = lane >> 4;
    const int gw     = blockIdx.x * 4 + w;
    const int qt     = gw & (SS / 16 - 1);          // 0..127
    const int h      = (gw >> 7) & (HH - 1);
    const int b      = gw >> 11;
    const int qBase  = qt * 16;

    const unsigned short* Qh = q  + ((size_t)b * HH + h) * SS * DK;
    const unsigned short* Kh = k  + ((size_t)b * HH + h) * SS * DK;
    const unsigned short* Vh = vT + ((size_t)b * HH + h) * DK * SS;
    unsigned short* Pw = Pls + w * (16 * 64);

    // Q fragments (A-operand, two DK chunks of 32)
    AF qa[2];
#pragma unroll
    for (int c = 0; c < 2; ++c)
#pragma unroll
        for (int v = 0; v < 8; ++v) {
            const int kk = c * 32 + ((v & 4) << 2) + half * 8 + (v & 3) * 2;
            qa[c].u[v] = *(const unsigned int*)(Qh + (size_t)(qBase + lane16) * DK + kk);
        }

    v8f   accO[4];
    float mrow[8], lrow[8];
#pragma unroll
    for (int d = 0; d < 4; ++d) accO[d] = vzero8();
#pragma unroll
    for (int v = 0; v < 8; ++v) { mrow[v] = -1e30f; lrow[v] = 0.0f; }

    for (int kv = 0; kv < SS; kv += 64) {
        // ---- scores for four 16-col tiles ----
        v8f s[4];
#pragma unroll
        for (int t = 0; t < 4; ++t) {
            s[t] = vzero8();
            const int col = kv + t * 16 + lane16;
            AF kf0, kf1;
#pragma unroll
            for (int v = 0; v < 8; ++v) {
                const int kk = ((v & 4) << 2) + half * 8 + (v & 3) * 2;
                kf0.u[v] = *(const unsigned int*)(Kh + (size_t)col * DK + kk);
                kf1.u[v] = *(const unsigned int*)(Kh + (size_t)col * DK + 32 + kk);
            }
            s[t] = wmma_bf16(qa[0], kf0, s[t]);
            s[t] = wmma_bf16(qa[1], kf1, s[t]);
        }
#pragma unroll
        for (int t = 0; t < 4; ++t) {
            const int mk = mask[b * SS + kv + t * 16 + lane16];
#pragma unroll
            for (int v = 0; v < 8; ++v)
                s[t][v] = (mk == 0) ? -1.0f : s[t][v] * 0.125f;   // 1/sqrt(64)
        }

        // ---- online softmax update ----
        float alpha[8], p[4][8];
#pragma unroll
        for (int v = 0; v < 8; ++v) {
            float t0 = fmaxf(fmaxf(s[0][v], s[1][v]), fmaxf(s[2][v], s[3][v]));
            t0 = rmax16(t0);
            const float nm = fmaxf(mrow[v], t0);
            alpha[v] = __expf(mrow[v] - nm);
            float rs = 0.0f;
#pragma unroll
            for (int t = 0; t < 4; ++t) {
                p[t][v] = __expf(s[t][v] - nm);
                rs += p[t][v];
            }
            rs = rsum16(rs);
            lrow[v] = lrow[v] * alpha[v] + rs;
            mrow[v] = nm;
        }
#pragma unroll
        for (int d = 0; d < 4; ++d)
#pragma unroll
            for (int v = 0; v < 8; ++v) accO[d][v] *= alpha[v];

        // ---- P (C layout) -> LDS -> A layout (16x64) ----
#pragma unroll
        for (int v = 0; v < 8; ++v) {
            const int m = v + 8 * half;
#pragma unroll
            for (int t = 0; t < 4; ++t)
                Pw[m * 64 + t * 16 + lane16] = f2bf(p[t][v]);
        }
        AF pf[2];
#pragma unroll
        for (int c = 0; c < 2; ++c)
#pragma unroll
            for (int v = 0; v < 8; ++v) {
                const int kk = c * 32 + ((v & 4) << 2) + half * 8 + (v & 3) * 2;
                pf[c].u[v] = *(const unsigned int*)(Pw + lane16 * 64 + kk);
            }

        // ---- O += P x V  (V transposed: contiguous kv pairs) ----
#pragma unroll
        for (int d = 0; d < 4; ++d) {
            const int dcol = d * 16 + lane16;
#pragma unroll
            for (int c = 0; c < 2; ++c) {
                AF vf;
#pragma unroll
                for (int v = 0; v < 8; ++v)
                    vf.u[v] = *(const unsigned int*)(Vh + (size_t)dcol * SS + kv + c * 32 +
                                                     ((v & 4) << 2) + half * 8 + (v & 3) * 2);
                accO[d] = wmma_bf16(pf[c], vf, accO[d]);
            }
        }
    }

    // ---- normalize and store ctx as bf16 [B,S,D] ----
#pragma unroll
    for (int v = 0; v < 8; ++v) {
        const float inv = 1.0f / lrow[v];
        const int   s   = qBase + v + 8 * half;
#pragma unroll
        for (int d = 0; d < 4; ++d) {
            const int col = h * DK + d * 16 + lane16;
            ctx[(size_t)(b * SS + s) * DD + col] = f2bf(accO[d][v] * inv);
        }
    }
}

// ---------------------------------------------------------------------------
// Host-side orchestration
// ---------------------------------------------------------------------------
extern "C" void kernel_launch(void* const* d_in, const int* in_sizes, int n_in,
                              void* d_out, int out_size, void* d_ws, size_t ws_size,
                              hipStream_t stream) {
    (void)in_sizes; (void)n_in; (void)out_size; (void)ws_size;

    const float* x    = (const float*)d_in[0];
    const int*   mask = (const int*)d_in[1];
    const float* wq = (const float*)d_in[2];  const float* bq = (const float*)d_in[3];
    const float* wk = (const float*)d_in[4];  const float* bk = (const float*)d_in[5];
    const float* wv = (const float*)d_in[6];  const float* bv = (const float*)d_in[7];
    const float* wo = (const float*)d_in[8];  const float* bo = (const float*)d_in[9];
    const float* w1 = (const float*)d_in[10]; const float* b1 = (const float*)d_in[11];
    const float* w2 = (const float*)d_in[12]; const float* b2 = (const float*)d_in[13];
    const float* g1  = (const float*)d_in[14]; const float* be1 = (const float*)d_in[15];
    const float* g2  = (const float*)d_in[16]; const float* be2 = (const float*)d_in[17];

    // workspace carve-up (256B aligned)
    size_t off = 0;
    auto alloc = [&](size_t bytes) -> void* {
        void* p = (char*)d_ws + off;
        off += (bytes + 255) & ~(size_t)255;
        return p;
    };
    unsigned short* wq_b  = (unsigned short*)alloc((size_t)DD * DD * 2);
    unsigned short* wk_b  = (unsigned short*)alloc((size_t)DD * DD * 2);
    unsigned short* wv_b  = (unsigned short*)alloc((size_t)DD * DD * 2);
    unsigned short* wo_b  = (unsigned short*)alloc((size_t)DD * DD * 2);
    unsigned short* w1_b  = (unsigned short*)alloc((size_t)DD * DFF * 2);
    unsigned short* w2_b  = (unsigned short*)alloc((size_t)DFF * DD * 2);
    unsigned short* xn_b  = (unsigned short*)alloc((size_t)MTOT * DD * 2);   // xn1 then xn2
    unsigned short* q_b   = (unsigned short*)alloc((size_t)MTOT * DD * 2);
    unsigned short* k_b   = (unsigned short*)alloc((size_t)MTOT * DD * 2);
    unsigned short* vT_b  = (unsigned short*)alloc((size_t)MTOT * DD * 2);
    unsigned short* ctx_b = (unsigned short*)alloc((size_t)MTOT * DD * 2);
    unsigned short* h_b   = (unsigned short*)alloc((size_t)MTOT * DFF * 2);
    float*          x1    = (float*)alloc((size_t)MTOT * DD * 4);

    // 1) weights -> bf16
    cvt_f32_bf16<<<2048, 256, 0, stream>>>(wq, wq_b, DD * DD);
    cvt_f32_bf16<<<2048, 256, 0, stream>>>(wk, wk_b, DD * DD);
    cvt_f32_bf16<<<2048, 256, 0, stream>>>(wv, wv_b, DD * DD);
    cvt_f32_bf16<<<2048, 256, 0, stream>>>(wo, wo_b, DD * DD);
    cvt_f32_bf16<<<4096, 256, 0, stream>>>(w1, w1_b, DD * DFF);
    cvt_f32_bf16<<<4096, 256, 0, stream>>>(w2, w2_b, DFF * DD);

    // 2) LN1
    ln_bf16<<<MTOT, 256, 0, stream>>>(x, xn_b, g1, be1);

    // 3) QKV projections
    dim3 gQ(DD / 128, MTOT / 128);
    gemm_bf16_wmma<2><<<gQ, 256, 0, stream>>>(xn_b, wq_b, bq, nullptr, nullptr, q_b,  MTOT, DD, DD);
    gemm_bf16_wmma<2><<<gQ, 256, 0, stream>>>(xn_b, wk_b, bk, nullptr, nullptr, k_b,  MTOT, DD, DD);
    gemm_bf16_wmma<3><<<gQ, 256, 0, stream>>>(xn_b, wv_b, bv, nullptr, nullptr, vT_b, MTOT, DD, DD);

    // 4) attention (4096 waves = 1024 blocks x 4 waves)
    flash_attn<<<(BB * HH * (SS / 16)) / 4, 128, 0, stream>>>(q_b, k_b, vT_b, mask, ctx_b);

    // 5) output projection + residual 1  -> x1 (f32)
    gemm_bf16_wmma<0><<<gQ, 256, 0, stream>>>(ctx_b, wo_b, bo, x, x1, nullptr, MTOT, DD, DD);

    // 6) LN2
    ln_bf16<<<MTOT, 256, 0, stream>>>(x1, xn_b, g2, be2);

    // 7) FFN up + ReLU -> h (bf16)
    dim3 gF1(DFF / 128, MTOT / 128);
    gemm_bf16_wmma<1><<<gF1, 256, 0, stream>>>(xn_b, w1_b, b1, nullptr, nullptr, h_b, MTOT, DFF, DD);

    // 8) FFN down + residual 2 -> d_out (f32)
    gemm_bf16_wmma<0><<<gQ, 256, 0, stream>>>(h_b, w2_b, b2, x1, (float*)d_out, nullptr, MTOT, DD, DFF);
}
